// ImageMLP_61229053771861
// MI455X (gfx1250) — matrix operand order repaired and checked by reading, not compiled, not run on hardware
//
#include <hip/hip_runtime.h>

typedef __attribute__((ext_vector_type(16))) _Float16 v16h;
typedef __attribute__((ext_vector_type(8)))  _Float16 v8h;
typedef __attribute__((ext_vector_type(2)))  _Float16 h2;
typedef __attribute__((ext_vector_type(8)))  float    v8f;

#define STRH 520               // LDS row stride in halves: 260 dwords = 4 mod 64 banks -> conflict-free b128 reads
#define ROWS_PER_WAVE 32
#define WAVES_PER_BLOCK 4
#define ROWS_PER_BLOCK (ROWS_PER_WAVE * WAVES_PER_BLOCK)
#define SMEM_BYTES (WAVES_PER_BLOCK * 2 * ROWS_PER_WAVE * STRH * 2)  // 266,240 B of 320KB WGP LDS

struct LayerP { const _Float16* w; const float* b; int K; int N; };
struct Params { LayerP L[7]; };

// ---------------------------------------------------------------------------
// Prep: convert f32 weights [K][N] into f16 B-fragment order for
// v_wmma_f32_16x16x32_f16.  Tile = 32(K) x 16(N) = 512 halves = 1024 B.
// Within a tile, lane l holds column n = l&15; lanes 0-15 hold K=0..15,
// lanes 16-31 hold K=16..31; half h of the 16-half fragment is K offset h.
// Tile order: t = n_tile * (K/32) + k_tile  (k-tiles contiguous for streaming).
// ---------------------------------------------------------------------------
__global__ void prep_weights(const float* __restrict__ W, _Float16* __restrict__ dst,
                             int K, int N) {
  int idx = blockIdx.x * blockDim.x + threadIdx.x;
  if (idx >= K * N) return;
  int t    = idx >> 9;
  int lane = (idx >> 4) & 31;
  int h    = idx & 15;
  int KT   = K >> 5;
  int kt   = t % KT;
  int nt   = t / KT;
  int k = kt * 32 + ((lane & 16) ? 16 : 0) + h;
  int n = nt * 16 + (lane & 15);
  dst[idx] = (_Float16)W[k * N + n];
}

// ---------------------------------------------------------------------------
// Fully-fused MLP: each wave32 pushes 32 rows through all 9 layers.
// Wave-private LDS ping-pong activation buffers; zero workgroup barriers.
// B-fragment loads use uniform SGPR base + divergent 32-bit lane offset (GVS).
// ---------------------------------------------------------------------------
__launch_bounds__(128, 1)
__global__ void mlp_fused(const float* __restrict__ x,
                          const float* __restrict__ W1, const float* __restrict__ b1,
                          const float* __restrict__ W9, const float* __restrict__ b9,
                          Params P, float* __restrict__ out, int nrows) {
  extern __shared__ _Float16 smem[];
  const int lane = threadIdx.x & 31;
  const int wid  = threadIdx.x >> 5;
  const int m0   = (blockIdx.x * WAVES_PER_BLOCK + wid) * ROWS_PER_WAVE;
  if (m0 >= nrows) return;

  _Float16* bufA = smem + (size_t)wid * (2 * ROWS_PER_WAVE * STRH);
  _Float16* bufB = bufA + ROWS_PER_WAVE * STRH;

  // ---- layer 1 (2 -> 128), VALU: one row per lane ----
  {
    int gr = m0 + lane;
    float x0 = x[gr * 2 + 0];
    float x1 = x[gr * 2 + 1];
    _Float16* row = bufA + lane * STRH;
    #pragma unroll 8
    for (int n = 0; n < 128; n += 2) {
      float h0v = fmaf(x0, W1[n],     fmaf(x1, W1[128 + n],     b1[n]));
      float h1v = fmaf(x0, W1[n + 1], fmaf(x1, W1[128 + n + 1], b1[n + 1]));
      h2 pk;
      pk[0] = (_Float16)fmaxf(h0v, 0.f);
      pk[1] = (_Float16)fmaxf(h1v, 0.f);
      *(h2*)(row + n) = pk;
    }
  }

  _Float16* inb  = bufA;
  _Float16* outb = bufB;

  const int arow = lane & 15;               // A-fragment: M = lane&15
  const int asel = (lane & 16) ? 8 : 0;     // upper half: A k offset asel<<1 = 16, C/D rows +8
  const int voff = lane * 32;               // divergent byte offset into a B tile

  // ---- layers 2..8: WMMA f16 -> f32 ----
  for (int li = 0; li < 7; ++li) {
    const int K = P.L[li].K;
    const int N = P.L[li].N;
    const _Float16* __restrict__ w    = P.L[li].w;
    const float*    __restrict__ bias = P.L[li].b;
    const int KT = K >> 5;

    for (int n0 = 0; n0 < N; n0 += 128) {          // 8 N-tiles per chunk
      v8f acc[2][8];
      #pragma unroll
      for (int j = 0; j < 8; ++j) {
        float bv = bias[n0 + j * 16 + arow];       // C/D: N = lane&15 per VGPR
        #pragma unroll
        for (int r = 0; r < 8; ++r) { acc[0][j][r] = bv; acc[1][j][r] = bv; }
      }

      // wave-uniform base for this 128-wide chunk (stays in SGPRs)
      const char* wchunk = (const char*)w + (size_t)(n0 >> 4) * KT * 1024;

      for (int kt = 0; kt < KT; ++kt) {
        // A fragments for the two 16-row stripes (contiguous k = 16*(lane/16) + s, same as B)
        const _Float16* a0p = inb + arow * STRH + kt * 32 + (asel << 1);
        v8h lo0 = *(const v8h*)a0p;
        v8h hi0 = *(const v8h*)(a0p + 8);
        const _Float16* a1p = a0p + 16 * STRH;
        v8h lo1 = *(const v8h*)a1p;
        v8h hi1 = *(const v8h*)(a1p + 8);
        v16h A0 = __builtin_shufflevector(lo0, hi0, 0,1,2,3,4,5,6,7,8,9,10,11,12,13,14,15);
        v16h A1 = __builtin_shufflevector(lo1, hi1, 0,1,2,3,4,5,6,7,8,9,10,11,12,13,14,15);

        #pragma unroll
        for (int j = 0; j < 8; ++j) {
          // uniform base (SGPR pair) + divergent 32-bit lane offset -> GVS form
          const char* bj = wchunk + ((size_t)j * KT + (size_t)kt) * 1024;
          v16h B = *(const v16h*)(bj + voff);
          acc[0][j] = __builtin_amdgcn_wmma_f32_16x16x32_f16(
              false, A0, false, B, (short)0, acc[0][j], false, false);
          acc[1][j] = __builtin_amdgcn_wmma_f32_16x16x32_f16(
              false, A1, false, B, (short)0, acc[1][j], false, false);
        }
      }

      // epilogue: ReLU, cvt f16, scatter per C/D layout (VGPR r -> row r / 8+r)
      #pragma unroll
      for (int s = 0; s < 2; ++s) {
        #pragma unroll
        for (int j = 0; j < 8; ++j) {
          int n  = n0 + j * 16 + arow;
          int mb = s * 16 + asel;
          #pragma unroll
          for (int r = 0; r < 8; ++r) {
            outb[(mb + r) * STRH + n] = (_Float16)fmaxf(acc[s][j][r], 0.f);
          }
        }
      }
    }
    _Float16* t = inb; inb = outb; outb = t;
  }

  // ---- layer 9 (128 -> 3) + sigmoid, VALU: one row per lane ----
  {
    const _Float16* row = inb + lane * STRH;
    float a0c = b9[0], a1c = b9[1], a2c = b9[2];
    #pragma unroll 4
    for (int k = 0; k < 128; k += 2) {
      h2 pk = *(const h2*)(row + k);
      float e0 = (float)pk[0], e1 = (float)pk[1];
      a0c = fmaf(e0, W9[k * 3 + 0], a0c); a0c = fmaf(e1, W9[(k + 1) * 3 + 0], a0c);
      a1c = fmaf(e0, W9[k * 3 + 1], a1c); a1c = fmaf(e1, W9[(k + 1) * 3 + 1], a1c);
      a2c = fmaf(e0, W9[k * 3 + 2], a2c); a2c = fmaf(e1, W9[(k + 1) * 3 + 2], a2c);
    }
    int gr = m0 + lane;
    out[gr * 3 + 0] = 1.f / (1.f + __expf(-a0c));
    out[gr * 3 + 1] = 1.f / (1.f + __expf(-a1c));
    out[gr * 3 + 2] = 1.f / (1.f + __expf(-a2c));
  }
}

// ---------------------------------------------------------------------------
extern "C" void kernel_launch(void* const* d_in, const int* in_sizes, int n_in,
                              void* d_out, int out_size, void* d_ws, size_t ws_size,
                              hipStream_t stream) {
  (void)n_in; (void)out_size; (void)ws_size;
  const float* x  = (const float*)d_in[0];
  const float* W1 = (const float*)d_in[1];
  const float* b1 = (const float*)d_in[2];
  const float* W9 = (const float*)d_in[17];
  const float* b9 = (const float*)d_in[18];

  static const int Ks[7] = {128, 256, 512, 512, 512, 512, 256};
  static const int Ns[7] = {256, 512, 512, 512, 512, 256, 128};

  _Float16* ws = (_Float16*)d_ws;
  Params P;
  size_t off = 0;
  for (int i = 0; i < 7; ++i) {
    const float* W = (const float*)d_in[3 + 2 * i];
    const float* b = (const float*)d_in[4 + 2 * i];
    int total = Ks[i] * Ns[i];
    prep_weights<<<(total + 255) / 256, 256, 0, stream>>>(W, ws + off, Ks[i], Ns[i]);
    P.L[i].w = ws + off;
    P.L[i].b = b;
    P.L[i].K = Ks[i];
    P.L[i].N = Ns[i];
    off += (size_t)total;
  }

  int nrows = in_sizes[0] / 2;
  hipFuncSetAttribute(reinterpret_cast<const void*>(mlp_fused),
                      hipFuncAttributeMaxDynamicSharedMemorySize, SMEM_BYTES);
  int blocks = (nrows + ROWS_PER_BLOCK - 1) / ROWS_PER_BLOCK;
  mlp_fused<<<blocks, 128, SMEM_BYTES, stream>>>(x, W1, b1, W9, b9, P,
                                                 (float*)d_out, nrows);
}
